// DLRMResNet_48576080117968
// MI455X (gfx1250) — compile-verified, matrix operand order checked by reading, not run on hardware
//
#include <hip/hip_runtime.h>
#include <hip/hip_bf16.h>

// ---------------------------------------------------------------------------
// DLRM forward for MI455X (gfx1250, wave32, WMMA bf16 16x16x32)
//   K1: fused bottom MLP 13->512->256->128 (ReLU), bf16 WMMA, wave-private
//   K2: async-gather (global_load_async_to_lds_b128) + 27x27 Gram via WMMA
//   K3: fused top MLP 506(->512)->256x4->1, accumulator-resident WMMA
// ---------------------------------------------------------------------------

typedef __bf16 bf16;
typedef __attribute__((ext_vector_type(16))) __bf16 v16bf;
typedef __attribute__((ext_vector_type(8)))  __bf16 v8bf;
typedef __attribute__((ext_vector_type(8)))  float  v8f;
typedef __attribute__((ext_vector_type(4)))  float  v4f;

#define VOCAB   2000000
#define PIECE   500000      // VOCAB / 4
#define BATCH   32768
#define XCOLS   39          // 13 dense + 26 sparse
#define NSP     26
#define ZLD     512         // padded top-input leading dim (506 valid)

// K2 per-wave LDS layout (bytes): comb bf16 32x128 = 8192, f32 stage 26x128 = 13312
#define K2_COMB_BYTES  (32 * 128 * 2)
#define K2_STAGE_BYTES (NSP * 128 * 4)
#define K2_WAVE_BYTES  (K2_COMB_BYTES + K2_STAGE_BYTES)

// ---- fragment loader: 16x32 (MxK) or (KxN via W row-major NxK) bf16 tile ---
// ISA 16-bit A layout: lanes 0-15 -> rows, K-halves {kb..kb+7, kb+16..kb+23},
// kb = (lane>>4)*8.  Two 16-byte vector loads per lane.
__device__ __forceinline__ v16bf load_frag(const bf16* __restrict__ base, int ld) {
  const int lane = threadIdx.x & 31;
  const bf16* p = base + (lane & 15) * ld + ((lane >> 4) << 3);
  v8bf lo = *(const v8bf*)(p);
  v8bf hi = *(const v8bf*)(p + 16);
  return __builtin_shufflevector(lo, hi, 0,1,2,3,4,5,6,7,8,9,10,11,12,13,14,15);
}

__device__ __forceinline__ v8f wmma_bf16(v16bf a, v16bf b, v8f c) {
  // D = A(16x32) * B(32x16) + C, f32 accumulate
  return __builtin_amdgcn_wmma_f32_16x16x32_bf16(false, a, false, b, (short)0, c,
                                                 false, false);
}

// C/D layout: n = n0 + (lane&15), m = (lane>>4)*8 + r.  bias + ReLU -> bf16
__device__ __forceinline__ void store_tile_relu(bf16* __restrict__ dst, int ld,
                                                v8f c, const float* __restrict__ bias,
                                                int n0) {
  const int lane = threadIdx.x & 31;
  const int n  = n0 + (lane & 15);
  const int mb = (lane >> 4) << 3;
  const float bv = bias[n];
#pragma unroll
  for (int r = 0; r < 8; ++r) {
    float v = c[r] + bv;
    dst[(mb + r) * ld + n] = (bf16)(v > 0.f ? v : 0.f);
  }
}

// ---- weight repack: f32 (rows x scol) -> bf16 (rows x dcol), zero-padded ---
__global__ void repack_bf16(const float* __restrict__ src, bf16* __restrict__ dst,
                            int rows, int scol, int dcol) {
  int i = blockIdx.x * blockDim.x + threadIdx.x;
  int total = rows * dcol;
  if (i >= total) return;
  int r = i / dcol, c = i - r * dcol;
  dst[i] = (bf16)(c < scol ? src[r * scol + c] : 0.f);
}

// ---- K1: fused bottom MLP, 4 waves/block, each wave owns 16 batch rows -----
__global__ void __launch_bounds__(128) bot_mlp(
    const float* __restrict__ x,
    const bf16* __restrict__ w0, const float* __restrict__ b0,   // 512 x 32(pad)
    const bf16* __restrict__ w1, const float* __restrict__ b1,   // 256 x 512
    const bf16* __restrict__ w2, const float* __restrict__ b2,   // 128 x 256
    bf16* __restrict__ hout) {                                   // BATCH x 128
  extern __shared__ __align__(16) char smem[];
  const int wave = threadIdx.x >> 5;
  const int lane = threadIdx.x & 31;
  bf16* a1 = (bf16*)smem + wave * (16 * 512 + 16 * 256);  // wave-private
  bf16* a2 = a1 + 16 * 512;
  const int row0 = (blockIdx.x * 4 + wave) * 16;

  // Input A-fragment (16x32, cols >= 13 zero-padded) straight from global x
  const int m  = row0 + (lane & 15);
  const int kb = (lane >> 4) << 3;
  v16bf afrag;
#pragma unroll
  for (int j = 0; j < 8; ++j) {
    int c0 = kb + j, c1 = kb + 16 + j;
    afrag[j]     = (bf16)(c0 < 13 ? x[(size_t)m * XCOLS + c0] : 0.f);
    afrag[8 + j] = (bf16)(c1 < 13 ? x[(size_t)m * XCOLS + c1] : 0.f);
  }

  // Layer 1: K=32 (single step), N=512
  for (int nt = 0; nt < 32; ++nt) {
    v8f acc = {};
    acc = wmma_bf16(afrag, load_frag(w0 + nt * 16 * 32, 32), acc);
    store_tile_relu(a1, 512, acc, b0, nt * 16);
  }

  // Layer 2: K=512, N=256, accumulator-resident
  {
    v8f acc[16] = {};
    for (int ks = 0; ks < 16; ++ks) {
      __builtin_prefetch(w1 + ((ks + 1) & 15) * 32, 0, 1);  // stream hint (L2-resident)
      v16bf a = load_frag(a1 + ks * 32, 512);
#pragma unroll
      for (int n = 0; n < 16; ++n)
        acc[n] = wmma_bf16(a, load_frag(w1 + (n * 16) * 512 + ks * 32, 512), acc[n]);
    }
#pragma unroll
    for (int n = 0; n < 16; ++n) store_tile_relu(a2, 256, acc[n], b1, n * 16);
  }

  // Layer 3: K=256, N=128 -> h (global, bf16)
  {
    v8f acc[8] = {};
    for (int ks = 0; ks < 8; ++ks) {
      v16bf a = load_frag(a2 + ks * 32, 256);
#pragma unroll
      for (int n = 0; n < 8; ++n)
        acc[n] = wmma_bf16(a, load_frag(w2 + (n * 16) * 256 + ks * 32, 256), acc[n]);
    }
#pragma unroll
    for (int n = 0; n < 8; ++n)
      store_tile_relu(hout + (size_t)row0 * 128, 128, acc[n], b2, n * 16);
  }
}

// ---- K2: async gather + interaction + z assembly, 1 sample per wave -------
__global__ void __launch_bounds__(128) interact(
    const float* __restrict__ x,
    const float* __restrict__ e0, const float* __restrict__ e1,
    const float* __restrict__ e2, const float* __restrict__ e3,
    const bf16* __restrict__ hbf, bf16* __restrict__ z) {
  extern __shared__ __align__(16) char smem[];
  const int wave = threadIdx.x >> 5;
  const int lane = threadIdx.x & 31;
  char* wbase = smem + wave * K2_WAVE_BYTES;
  bf16*  comb  = (bf16*)wbase;                       // 32 rows x 128 bf16
  float* stage = (float*)(wbase + K2_COMB_BYTES);    // 26 rows x 128 f32
  // workgroup-relative LDS byte offset of this wave's f32 staging region
  // (no static __shared__ in this kernel -> dynamic segment starts at 0)
  const unsigned stage_lds = (unsigned)(wave * K2_WAVE_BYTES + K2_COMB_BYTES);
  const int s = blockIdx.x * 4 + wave;

  // fetch all 26 indices with one predicated lane load, broadcast via shfl
  int myidx = 0;
  if (lane < NSP) myidx = ((int)x[(size_t)s * XCOLS + 13 + lane]) % VOCAB;

  // rows 1..26: async DMA each 512B embedding row straight into LDS.
  // One global_load_async_to_lds_b128 per row (32 lanes x 16B), ASYNCcnt-tracked:
  // all 26 scattered HBM fetches are in flight at once, no VGPR round-trip.
  for (int f = 0; f < NSP; ++f) {
    int idx = __shfl(myidx, f, 32);
    int piece = idx / PIECE;
    const float* tab = piece == 0 ? e0 : piece == 1 ? e1 : piece == 2 ? e2 : e3;
    const float* rowp = tab + (size_t)(idx - piece * PIECE) * 128;
    unsigned long long ga = (unsigned long long)(rowp + lane * 4);
    unsigned la = stage_lds + (unsigned)(f * 512 + lane * 16);
    asm volatile("global_load_async_to_lds_b128 %0, %1, off"
                 :: "v"(la), "v"(ga) : "memory");
  }

  // row 0 = h (overlaps with async fetches)
#pragma unroll
  for (int j = 0; j < 4; ++j)
    comb[lane * 4 + j] = hbf[(size_t)s * 128 + lane * 4 + j];
  // zero rows 27..31 (uniform trip count: 640/32 = 20)
  for (int i = lane; i < 5 * 128; i += 32) comb[27 * 128 + i] = (bf16)0.f;

  bf16* zrow = z + (size_t)s * ZLD;
#pragma unroll
  for (int j = 0; j < 4; ++j) zrow[lane * 4 + j] = comb[lane * 4 + j];
  if (lane < 6) zrow[506 + lane] = (bf16)0.f;

  // wait for the 26 async row DMAs, then convert f32 stage -> bf16 comb
  asm volatile("s_wait_asynccnt 0x0" ::: "memory");
  for (int t = 0; t < NSP; ++t) {
    v4f v = *(const v4f*)(stage + t * 128 + lane * 4);
    bf16* d = comb + (1 + t) * 128 + lane * 4;
    d[0] = (bf16)v[0]; d[1] = (bf16)v[1]; d[2] = (bf16)v[2]; d[3] = (bf16)v[3];
  }

  // Gram = comb * comb^T over K=128; only tiles with m<=n needed for triu
  const int n15 = lane & 15, hi = lane >> 4;
#pragma unroll
  for (int mt = 0; mt < 2; ++mt) {
#pragma unroll
    for (int nt = mt; nt < 2; ++nt) {
      v8f acc = {};
#pragma unroll
      for (int ks = 0; ks < 4; ++ks) {
        v16bf a = load_frag(comb + mt * 16 * 128 + ks * 32, 128);
        v16bf b = load_frag(comb + nt * 16 * 128 + ks * 32, 128);
        acc = wmma_bf16(a, b, acc);
      }
      int n = nt * 16 + n15;
#pragma unroll
      for (int r = 0; r < 8; ++r) {
        int m = mt * 16 + hi * 8 + r;
        if (m <= n && n < 27) {
          int flat = 27 * m - (m * (m - 1)) / 2 + (n - m);
          zrow[128 + flat] = (bf16)acc[r];
        }
      }
    }
  }
}

// ---- K3: fused top MLP, 4 waves/block, 16 rows per wave --------------------
__global__ void __launch_bounds__(128) top_mlp(
    const bf16* __restrict__ z,
    const bf16* __restrict__ w0, const float* __restrict__ b0,  // 256 x 512(pad)
    const bf16* __restrict__ w1, const float* __restrict__ b1,  // 256 x 256
    const bf16* __restrict__ w2, const float* __restrict__ b2,
    const bf16* __restrict__ w3, const float* __restrict__ b3,
    const float* __restrict__ w4, const float* __restrict__ b4, // 1 x 256 (f32)
    float* __restrict__ out) {
  extern __shared__ __align__(16) char smem[];
  const int wave = threadIdx.x >> 5;
  const int lane = threadIdx.x & 31;
  bf16* act0 = (bf16*)smem + wave * (2 * 16 * 256);
  bf16* act1 = act0 + 16 * 256;
  const int row0 = (blockIdx.x * 4 + wave) * 16;

  // t0: K=512 (A frags direct from global z, one load per k-step), N=256
  {
    v8f acc[16] = {};
    for (int ks = 0; ks < 16; ++ks) {
      __builtin_prefetch(z + (size_t)row0 * ZLD + ((ks + 1) & 15) * 32, 0, 1);
      v16bf a = load_frag(z + (size_t)row0 * ZLD + ks * 32, ZLD);
#pragma unroll
      for (int n = 0; n < 16; ++n)
        acc[n] = wmma_bf16(a, load_frag(w0 + (n * 16) * 512 + ks * 32, 512), acc[n]);
    }
#pragma unroll
    for (int n = 0; n < 16; ++n) store_tile_relu(act0, 256, acc[n], b0, n * 16);
  }

  // t1..t3: K=256, N=256, ping-pong in LDS
  const bf16* ws_[3] = {w1, w2, w3};
  const float* bs_[3] = {b1, b2, b3};
  bf16* src = act0;
  bf16* dst = act1;
#pragma unroll
  for (int L = 0; L < 3; ++L) {
    v8f acc[16] = {};
    for (int ks = 0; ks < 8; ++ks) {
      v16bf a = load_frag(src + ks * 32, 256);
#pragma unroll
      for (int n = 0; n < 16; ++n)
        acc[n] = wmma_bf16(a, load_frag(ws_[L] + (n * 16) * 256 + ks * 32, 256), acc[n]);
    }
#pragma unroll
    for (int n = 0; n < 16; ++n) store_tile_relu(dst, 256, acc[n], bs_[L], n * 16);
    bf16* t = src; src = dst; dst = t;
  }

  // t4: 256 -> 1, VALU dot + cross-half shuffle reduce
  {
    const int r = lane & 15, kh = lane >> 4;
    float sum = 0.f;
    const bf16* arow = src + r * 256 + kh * 128;
    const float* wrow = w4 + kh * 128;
    for (int k = 0; k < 128; ++k) sum += (float)arow[k] * wrow[k];
    sum += __shfl_xor(sum, 16, 32);
    if (kh == 0) out[row0 + r] = sum + b4[0];
  }
}

// ---------------------------------------------------------------------------
extern "C" void kernel_launch(void* const* d_in, const int* in_sizes, int n_in,
                              void* d_out, int out_size, void* d_ws, size_t ws_size,
                              hipStream_t stream) {
  const float* x    = (const float*)d_in[0];
  const float* e0   = (const float*)d_in[1];
  const float* e1   = (const float*)d_in[2];
  const float* e2   = (const float*)d_in[3];
  const float* e3   = (const float*)d_in[4];
  const float* bw0  = (const float*)d_in[5];
  const float* bb0  = (const float*)d_in[6];
  const float* bw1  = (const float*)d_in[7];
  const float* bb1  = (const float*)d_in[8];
  const float* bw2  = (const float*)d_in[9];
  const float* bb2  = (const float*)d_in[10];
  const float* tw0  = (const float*)d_in[11];
  const float* tb0  = (const float*)d_in[12];
  const float* tw1  = (const float*)d_in[13];
  const float* tb1  = (const float*)d_in[14];
  const float* tw2  = (const float*)d_in[15];
  const float* tb2  = (const float*)d_in[16];
  const float* tw3  = (const float*)d_in[17];
  const float* tb3  = (const float*)d_in[18];
  const float* tw4  = (const float*)d_in[19];
  const float* tb4  = (const float*)d_in[20];
  float* out = (float*)d_out;

  // workspace carve (bytes)
  char* ws = (char*)d_ws;
  size_t off = 0;
  auto carve = [&](size_t bytes) { char* p = ws + off; off += (bytes + 255) & ~(size_t)255; return p; };
  bf16* w0p = (bf16*)carve(512 * 32 * 2);        // bw0 padded K 13->32
  bf16* w1p = (bf16*)carve(256 * 512 * 2);
  bf16* w2p = (bf16*)carve(128 * 256 * 2);
  bf16* t0p = (bf16*)carve(256 * 512 * 2);       // tw0 padded K 506->512
  bf16* t1p = (bf16*)carve(256 * 256 * 2);
  bf16* t2p = (bf16*)carve(256 * 256 * 2);
  bf16* t3p = (bf16*)carve(256 * 256 * 2);
  bf16* hbf = (bf16*)carve((size_t)BATCH * 128 * 2);
  bf16* zbf = (bf16*)carve((size_t)BATCH * ZLD * 2);
  (void)ws_size; (void)n_in; (void)in_sizes; (void)out_size;

  auto rp = [&](const float* s, bf16* d, int r, int sc, int dc) {
    int tot = r * dc;
    repack_bf16<<<(tot + 255) / 256, 256, 0, stream>>>(s, d, r, sc, dc);
  };
  rp(bw0, w0p, 512, 13, 32);
  rp(bw1, w1p, 256, 512, 512);
  rp(bw2, w2p, 128, 256, 256);
  rp(tw0, t0p, 256, 506, 512);
  rp(tw1, t1p, 256, 256, 256);
  rp(tw2, t2p, 256, 256, 256);
  rp(tw3, t3p, 256, 256, 256);

  // K1: bottom MLP — 64 rows/block (4 waves x 16)
  bot_mlp<<<BATCH / 64, 128, 4 * (16 * 512 + 16 * 256) * sizeof(bf16), stream>>>(
      x, w0p, bb0, w1p, bb1, w2p, bb2, hbf);

  // K2: async gather + interaction — 4 samples/block
  interact<<<BATCH / 4, 128, 4 * K2_WAVE_BYTES, stream>>>(
      x, e0, e1, e2, e3, hbf, zbf);

  // K3: top MLP — 64 rows/block
  top_mlp<<<BATCH / 64, 128, 4 * (2 * 16 * 256) * sizeof(bf16), stream>>>(
      zbf, t0p, tb0, t1p, tb1, t2p, tb2, t3p, tb3, tw4, tb4, out);
}
